// Panel_22196390985827
// MI455X (gfx1250) — matrix operand order repaired and checked by reading, not compiled, not run on hardware
//
#include <hip/hip_runtime.h>
#include <hip/hip_bf16.h>

typedef __attribute__((ext_vector_type(16))) __bf16 v16bf;
typedef __attribute__((ext_vector_type(8)))  float  v8f;

#define WMMA_BF16(A, B, C) \
  __builtin_amdgcn_wmma_f32_16x16x32_bf16(false, (A), false, (B), (short)0, (C), false, false)

// ---------- helpers ----------

__device__ __forceinline__ unsigned short f2bf(float x) {
  union { float f; unsigned int u; } v; v.f = x;
  unsigned int r = v.u + 0x7FFFu + ((v.u >> 16) & 1u);   // RNE
  return (unsigned short)(r >> 16);
}

__device__ __forceinline__ unsigned int f2bf2(float lo, float hi) {
  return (unsigned int)f2bf(lo) | ((unsigned int)f2bf(hi) << 16);
}

// A-matrix fragment (16x32, 16-bit): lane l (0-15) row M=l, K=0..7,16..23;
// lanes 16-31 row M=l-16, K=8..15,24..31. Pairs packed per VGPR.
__device__ __forceinline__ v16bf load_fragA(const unsigned short* base, int stride, int lane) {
  int row  = lane & 15;
  int koff = (lane & 16) ? 8 : 0;
  const unsigned int* p = (const unsigned int*)(base + (size_t)row * stride);
  union { v16bf v; unsigned int u[8]; } f;
#pragma unroll
  for (int i = 0; i < 8; ++i) {
    int k0 = ((i < 4) ? (2 * i) : (16 + 2 * (i - 4))) + koff;
    f.u[i] = p[k0 >> 1];
  }
  return f.v;
}

// B-matrix fragment (32x16, 16-bit), sourced from B^T (N-major rows, K contiguous):
// lane l holds column N=l&15; lanes 0-15: K=0..7,16..23, lanes 16-31: K=8..15,24..31 (same as A).
__device__ __forceinline__ v16bf load_fragB(const unsigned short* baseT, int stride, int lane) {
  int col  = lane & 15;
  int koff = (lane & 16) ? 8 : 0;
  const unsigned int* p = (const unsigned int*)(baseT + (size_t)col * stride);
  union { v16bf v; unsigned int u[8]; } f;
#pragma unroll
  for (int i = 0; i < 8; ++i) f.u[i] = p[(((i < 4) ? (2 * i) : (16 + 2 * (i - 4))) + koff) >> 1];
  return f.v;
}

// ---------- kernel 1/3/5: generic 64x64 @ 64xN GEMM per batch ----------
// Out = W(64x64) @ (X + Add)  , X/Add: (B, 64, N) f32
// MODE 0: OutF f32 (B,64,N); MODE 1: OutB bf16 transposed (B,N,64); MODE 2: OutB bf16 (B,64,N)
template <int MODE, bool HASADD>
__global__ void gemm64_kernel(const float* __restrict__ W, const float* __restrict__ X,
                              const float* __restrict__ Add, float* __restrict__ OutF,
                              unsigned short* __restrict__ OutB, int N) {
  int b   = blockIdx.y;
  int px0 = blockIdx.x * 64;
  int tid = threadIdx.x;                 // 256 threads = 8 waves

  __shared__ __align__(16) unsigned short Wl[64 * 64];   // (o,c) row-major
  __shared__ __align__(16) unsigned short Xl[64 * 64];   // B^T: (n,c) row-major

#pragma unroll
  for (int i = 0; i < 8; ++i) {
    int e = tid * 8 + i;                                  // 2048 per half; 2 rounds
    Wl[e] = f2bf(W[e]);
    Wl[e + 2048] = f2bf(W[e + 2048]);
  }
  {
    const float* xb = X + ((size_t)b * 64) * N + px0;
    const float* ab = HASADD ? Add + ((size_t)b * 64) * N + px0 : nullptr;
    int c  = tid >> 2;
    int nb = (tid & 3) * 16;
#pragma unroll
    for (int i = 0; i < 16; ++i) {
      int n = nb + i;
      float v = xb[(size_t)c * N + n];
      if (HASADD) v += ab[(size_t)c * N + n];
      Xl[n * 64 + c] = f2bf(v);
    }
  }
  __syncthreads();

  int wave = tid >> 5, lane = tid & 31;
  int n = lane & 15, mhi = (lane & 16) ? 8 : 0;
#pragma unroll
  for (int t = wave * 2; t < wave * 2 + 2; ++t) {
    int mt = t >> 2, nt = t & 3;
    v8f acc = {0.f, 0.f, 0.f, 0.f, 0.f, 0.f, 0.f, 0.f};
#pragma unroll
    for (int kt = 0; kt < 2; ++kt) {
      v16bf a  = load_fragA(Wl + mt * 16 * 64 + kt * 32, 64, lane);
      v16bf bb = load_fragB(Xl + nt * 16 * 64 + kt * 32, 64, lane);
      acc = WMMA_BF16(a, bb, acc);
    }
    int px = px0 + nt * 16 + n;
    if (MODE == 0) {
#pragma unroll
      for (int r = 0; r < 8; ++r)
        OutF[((size_t)b * 64 + mt * 16 + r + mhi) * N + px] = acc[r];
    } else if (MODE == 1) {
      // o is contiguous over r -> pack pairs, 4x b32 stores
      unsigned int* po =
          (unsigned int*)(OutB + ((size_t)b * N + px) * 64 + mt * 16 + mhi);
#pragma unroll
      for (int r = 0; r < 4; ++r) po[r] = f2bf2(acc[2 * r], acc[2 * r + 1]);
    } else {
#pragma unroll
      for (int r = 0; r < 8; ++r)
        OutB[((size_t)b * 64 + mt * 16 + r + mhi) * N + px] = f2bf(acc[r]);
    }
  }
}

// ---------- kernel 2: fused pixel_unshuffle + depthwise 3x3 + channel split ----------
// c_out: (B,64,64,64) f32 ; Wdw: (256,1,3,3) ; sout4: j=0..3 planes of (B,64,32,32)
__global__ void dwconv_kernel(const float* __restrict__ c_out, const float* __restrict__ Wdw,
                              float* __restrict__ sout4) {
  int j   = blockIdx.y;                            // 0:s_x 1:s_q 2:s_k 3:s_v
  int idx = blockIdx.x * 256 + threadIdx.x;        // over B*64*1024
  int w = idx & 31, h = (idx >> 5) & 31, c = (idx >> 10) & 63, b = idx >> 16;
  int jr = j >> 1, jc = j & 1;
  const float* src = c_out + ((size_t)b * 64 + c) * 4096;
  const float* wk  = Wdw + (size_t)(4 * c + j) * 9;
  float acc = 0.f;
#pragma unroll
  for (int dy = -1; dy <= 1; ++dy) {
    int hy = h + dy;
    if (hy < 0 || hy >= 32) continue;
    int sy = 2 * hy + jr;
#pragma unroll
    for (int dx = -1; dx <= 1; ++dx) {
      int wx = w + dx;
      if (wx < 0 || wx >= 32) continue;
      acc += wk[(dy + 1) * 3 + (dx + 1)] * src[sy * 64 + 2 * wx + jc];
    }
  }
  sout4[(size_t)j * 2097152 + idx] = acc;
}

// ---------- kernel 4: attention (softmax(q^T k) @ v^T), one block = 16 query rows ----------
// qT,kT: (B, L=1024, 64) bf16 ; vbf: (B, 64, L) bf16 ; qkv: (B, 64, L) f32
__global__ void attn_kernel(const unsigned short* __restrict__ qT,
                            const unsigned short* __restrict__ kT,
                            const unsigned short* __restrict__ vbf,
                            float* __restrict__ qkv) {
  const int L = 1024;
  int b  = blockIdx.y;
  int l0 = blockIdx.x * 16;
  int tid = threadIdx.x;                  // 128 threads = 4 waves
  int wave = tid >> 5, lane = tid & 31;
  int n = lane & 15, mhi = (lane & 16) ? 8 : 0;

  __shared__ __align__(16) float          Ssc[16][1024];   // 64 KB scores
  __shared__ __align__(16) unsigned short Pb[16][1024];    // 32 KB probs (bf16)
  __shared__ float red[16][8];
  __shared__ float rowmax[16], rowsum[16];

  // ---- phase 1: S = q^T k  (M=l tile of 16, N=m, K=c=64) ----
  const unsigned short* qbase = qT + ((size_t)b * L + l0) * 64;
  v16bf a0 = load_fragA(qbase, 64, lane);
  v16bf a1 = load_fragA(qbase + 32, 64, lane);
  const unsigned short* kbatch = kT + (size_t)b * L * 64;
#pragma unroll 2
  for (int mt = wave; mt < 64; mt += 4) {
    int mtn = (mt + 4 < 64) ? mt + 4 : mt;                 // prefetch next tile
    __builtin_prefetch(kbatch + (size_t)mtn * 16 * 64, 0, 1);
    v8f acc = {0.f, 0.f, 0.f, 0.f, 0.f, 0.f, 0.f, 0.f};
    v16bf b0 = load_fragB(kbatch + (size_t)mt * 16 * 64, 64, lane);
    v16bf b1 = load_fragB(kbatch + (size_t)mt * 16 * 64 + 32, 64, lane);
    acc = WMMA_BF16(a0, b0, acc);
    acc = WMMA_BF16(a1, b1, acc);
#pragma unroll
    for (int r = 0; r < 8; ++r) Ssc[r + mhi][mt * 16 + n] = acc[r];
  }
  __syncthreads();

  // ---- phase 2: row softmax over 1024 (8 threads per row) ----
  int row = tid >> 3, sub = tid & 7;
  {
    float mx = -3.4e38f;
    for (int jx = sub * 128; jx < sub * 128 + 128; ++jx) mx = fmaxf(mx, Ssc[row][jx]);
    red[row][sub] = mx;
  }
  __syncthreads();
  if (sub == 0) {
    float m2 = red[row][0];
#pragma unroll
    for (int i = 1; i < 8; ++i) m2 = fmaxf(m2, red[row][i]);
    rowmax[row] = m2;
  }
  __syncthreads();
  {
    float rm = rowmax[row], sm = 0.f;
    for (int jx = sub * 128; jx < sub * 128 + 128; ++jx) {
      float e = __expf(Ssc[row][jx] - rm);
      Ssc[row][jx] = e;
      sm += e;
    }
    red[row][sub] = sm;
  }
  __syncthreads();
  if (sub == 0) {
    float s2 = 0.f;
#pragma unroll
    for (int i = 0; i < 8; ++i) s2 += red[row][i];
    rowsum[row] = s2;
  }
  __syncthreads();
  {
    float inv = 1.0f / rowsum[row];
    for (int jx = sub * 128; jx < sub * 128 + 128; jx += 2) {
      unsigned int pk = f2bf2(Ssc[row][jx] * inv, Ssc[row][jx + 1] * inv);
      *(unsigned int*)&Pb[row][jx] = pk;
    }
  }
  __syncthreads();

  // ---- phase 3: out[l,c] = P @ v^T  (M=l, N=c tile per wave, K=m=1024) ----
  int ct = wave;
  v8f acc = {0.f, 0.f, 0.f, 0.f, 0.f, 0.f, 0.f, 0.f};
  const unsigned short* vb = vbf + ((size_t)b * 64 + ct * 16) * (size_t)L;
#pragma unroll 4
  for (int kt = 0; kt < 32; ++kt) {
    int ktn = (kt + 4 < 32) ? kt + 4 : kt;
    __builtin_prefetch(vb + ktn * 32, 0, 1);
    v16bf a  = load_fragA(&Pb[0][0] + kt * 32, 1024, lane);
    v16bf bb = load_fragB(vb + kt * 32, 1024, lane);
    acc = WMMA_BF16(a, bb, acc);
  }
  // l is contiguous over r -> two float4 (b128) stores
  float* po = qkv + ((size_t)b * 64 + ct * 16 + n) * L + l0 + mhi;
  float4 lo = make_float4(acc[0], acc[1], acc[2], acc[3]);
  float4 hi = make_float4(acc[4], acc[5], acc[6], acc[7]);
  *(float4*)(po)     = lo;
  *(float4*)(po + 4) = hi;
}

// ---------- kernels 6/7: deterministic per-batch mean/var ----------
__global__ void reduce_partial_kernel(const float* __restrict__ s_out, float* __restrict__ partial) {
  int b = blockIdx.y, chunk = blockIdx.x, tid = threadIdx.x;
  const float* p = s_out + (size_t)b * 65536 + chunk * 1024;
  float s1 = 0.f, s2 = 0.f;
  for (int i = tid; i < 1024; i += 256) { float v = p[i]; s1 += v; s2 += v * v; }
  __shared__ float r1[256], r2[256];
  r1[tid] = s1; r2[tid] = s2;
  __syncthreads();
  for (int s = 128; s > 0; s >>= 1) {
    if (tid < s) { r1[tid] += r1[tid + s]; r2[tid] += r2[tid + s]; }
    __syncthreads();
  }
  if (tid == 0) {
    partial[(b * 64 + chunk) * 2]     = r1[0];
    partial[(b * 64 + chunk) * 2 + 1] = r2[0];
  }
}

__global__ void finalize_stats_kernel(const float* __restrict__ partial, float* __restrict__ stats) {
  int b = blockIdx.x, tid = threadIdx.x;   // 64 threads
  __shared__ float r1[64], r2[64];
  r1[tid] = partial[(b * 64 + tid) * 2];
  r2[tid] = partial[(b * 64 + tid) * 2 + 1];
  __syncthreads();
  for (int s = 32; s > 0; s >>= 1) {
    if (tid < s) { r1[tid] += r1[tid + s]; r2[tid] += r2[tid + s]; }
    __syncthreads();
  }
  if (tid == 0) {
    float mean = r1[0] * (1.0f / 65536.0f);
    float var  = r2[0] * (1.0f / 65536.0f) - mean * mean;
    stats[2 * b]     = mean;
    stats[2 * b + 1] = rsqrtf(var + 1e-5f);
  }
}

// ---------- kernel 8: GN/BN affine + PReLU + pixel_shuffle + residual ----------
__global__ void epilogue_kernel(const float* __restrict__ s_out, const float* __restrict__ sc_x,
                                const float* __restrict__ gn_g, const float* __restrict__ gn_b,
                                const float* __restrict__ bn_g, const float* __restrict__ bn_b,
                                const float* __restrict__ prelu_a, const float* __restrict__ stats,
                                float* __restrict__ out) {
  int idx = blockIdx.x * 256 + threadIdx.x;     // B*64*64*64 = 8388608
  int b   = idx >> 18;
  int rem = idx & 262143;
  int co  = rem >> 12;
  int y   = (rem >> 6) & 63;
  int x   = rem & 63;
  int ch  = co * 4 + (y & 1) * 2 + (x & 1);     // pixel_shuffle source channel (0..255)
  int c   = ch & 63;                            // 4x concat -> single-copy channel
  int h = y >> 1, w = x >> 1;
  float v = s_out[((size_t)b * 64 + c) * 1024 + h * 32 + w];
  v = (v - stats[2 * b]) * stats[2 * b + 1];
  v = v * gn_g[ch] + gn_b[ch];
  v = v * rsqrtf(1.0f + 1e-5f) * bn_g[ch] + bn_b[ch];
  float a = prelu_a[0];
  v = (v >= 0.f) ? v : a * v;
  out[idx] = sc_x[idx] + v;
}

// ---------- orchestration ----------
extern "C" void kernel_launch(void* const* d_in, const int* in_sizes, int n_in,
                              void* d_out, int out_size, void* d_ws, size_t ws_size,
                              hipStream_t stream) {
  (void)in_sizes; (void)n_in; (void)out_size; (void)ws_size;
  const float* x    = (const float*)d_in[0];
  const float* sc_x = (const float*)d_in[1];
  const float* Wc   = (const float*)d_in[2];
  const float* Wdw  = (const float*)d_in[3];
  const float* Wq   = (const float*)d_in[4];
  const float* Wk   = (const float*)d_in[5];
  const float* Wv   = (const float*)d_in[6];
  const float* Wx   = (const float*)d_in[7];
  const float* gn_g = (const float*)d_in[8];
  const float* gn_b = (const float*)d_in[9];
  const float* bn_g = (const float*)d_in[10];
  const float* bn_b = (const float*)d_in[11];
  const float* pa   = (const float*)d_in[12];
  float* out = (float*)d_out;

  const size_t NPB = 2097152;   // 32*64*1024 elems per (B,64,1024) buffer
  char* ws = (char*)d_ws;
  float* c_out = (float*)ws;                 ws += (size_t)8388608 * 4;   // (B,64,4096)
  float* sPack = (float*)ws;                 ws += 4 * NPB * 4;           // s_x,s_q,s_k,s_v
  float* s_x = sPack;
  float* s_q = sPack + NPB;
  float* s_k = sPack + 2 * NPB;
  float* s_v = sPack + 3 * NPB;
  unsigned short* qT  = (unsigned short*)ws; ws += NPB * 2;               // (B,L,64) bf16
  unsigned short* kT  = (unsigned short*)ws; ws += NPB * 2;               // (B,L,64) bf16
  unsigned short* vbf = (unsigned short*)ws; ws += NPB * 2;               // (B,64,L) bf16
  float* qkv   = (float*)ws;                 ws += NPB * 4;               // (B,64,L)
  float* s_out = (float*)ws;                 ws += NPB * 4;               // (B,64,1024)
  float* partial = (float*)ws;               ws += (size_t)32 * 64 * 2 * 4;
  float* stats = (float*)ws;                 ws += 64 * 4;

  // 1) c_out = Wc @ x
  gemm64_kernel<0, false><<<dim3(64, 32), 256, 0, stream>>>(Wc, x, nullptr, c_out, nullptr, 4096);
  // 2) unshuffle + depthwise 3x3 + split
  dwconv_kernel<<<dim3(8192, 4), 256, 0, stream>>>(c_out, Wdw, sPack);
  // 3) q/k/v projections -> bf16 (q,k transposed for WMMA A/B fragments)
  gemm64_kernel<1, false><<<dim3(16, 32), 256, 0, stream>>>(Wq, s_q, nullptr, nullptr, qT, 1024);
  gemm64_kernel<1, false><<<dim3(16, 32), 256, 0, stream>>>(Wk, s_k, nullptr, nullptr, kT, 1024);
  gemm64_kernel<2, false><<<dim3(16, 32), 256, 0, stream>>>(Wv, s_v, nullptr, nullptr, vbf, 1024);
  // 4) attention
  attn_kernel<<<dim3(64, 32), 128, 0, stream>>>(qT, kT, vbf, qkv);
  // 5) s_out = Wx @ (s_x + qkv)
  gemm64_kernel<0, true><<<dim3(16, 32), 256, 0, stream>>>(Wx, s_x, qkv, s_out, nullptr, 1024);
  // 6-7) per-batch mean / var (deterministic 2-stage)
  reduce_partial_kernel<<<dim3(64, 32), 256, 0, stream>>>(s_out, partial);
  finalize_stats_kernel<<<32, 64, 0, stream>>>(partial, stats);
  // 8) norm + affine + PReLU + pixel_shuffle + residual
  epilogue_kernel<<<32768, 256, 0, stream>>>(s_out, sc_x, gn_g, gn_b, bn_g, bn_b, pa, stats, out);
}
